// DGCNN_11965778887303
// MI455X (gfx1250) — matrix operand order repaired and checked by reading, not compiled, not run on hardware
//
#include <hip/hip_runtime.h>
#include <stdint.h>

// ---------------------------------------------------------------------------
// DGCNN part-segmentation forward for MI455X (gfx1250, wave32, WMMA + TDM).
// All contractions use v_wmma_f32_16x16x32_f16; activations kept in f16
// point-major (n, c) layout so WMMA fragments are contiguous loads and the
// channel concats are free column slices of one buffer. Pointwise GEMMs stage
// the 16-point activation tile into LDS with the Tensor Data Mover and share
// it across 4 waves (4x less activation traffic).
// ---------------------------------------------------------------------------

typedef __attribute__((ext_vector_type(16))) _Float16 v16h;
typedef __attribute__((ext_vector_type(8)))  _Float16 v8h;
typedef __attribute__((ext_vector_type(8)))  float    v8f;
typedef __attribute__((ext_vector_type(4)))  unsigned int u32x4;
typedef __attribute__((ext_vector_type(8)))  int i32x8;
typedef __attribute__((ext_vector_type(4)))  int i32x4;

#define KNN 20

#if defined(__gfx1250__) && __has_builtin(__builtin_amdgcn_tensor_load_to_lds) && \
    __has_builtin(__builtin_amdgcn_s_wait_tensorcnt)
#define USE_TDM 1
#else
#define USE_TDM 0
#endif

__device__ __forceinline__ v8f wmma16(v16h a, v16h b, v8f c) {
  // D = A(16x32 f16) x B(32x16 f16) + C(16x16 f32)
  return __builtin_amdgcn_wmma_f32_16x16x32_f16(
      false, a, false, b, (short)0, c, false, false);
}

// A-fragment (16x32 f16, M x K): lane L holds row M = L%16;
// elements [0..15] = A[M][k0 .. k0+15],  k0 = (L<16 ? 0 : 16)
// base = &A[M][kchunk*32]
__device__ __forceinline__ v16h load_a16(const _Float16* base, int half) {
  const _Float16* p = base + (half ? 16 : 0);
  v8h lo = *(const v8h*)(p);
  v8h hi = *(const v8h*)(p + 8);
  v16h r;
#pragma unroll
  for (int i = 0; i < 8; ++i) { r[i] = lo[i]; r[i + 8] = hi[i]; }
  return r;
}

// B-fragment (32x16 f16, K x N): lane L holds col N = L%16, 16 consecutive
// K values starting at (L<16 ? 0 : 16) within the k-chunk. p = &X[N][kbase].
// Loaded as 2x16B so only 16-byte alignment is required (LDS or global).
__device__ __forceinline__ v16h load_b16(const _Float16* p) {
  v8h lo = *(const v8h*)(p);
  v8h hi = *(const v8h*)(p + 8);
  v16h r;
#pragma unroll
  for (int i = 0; i < 8; ++i) { r[i] = lo[i]; r[i + 8] = hi[i]; }
  return r;
}

// top-k (smallest distance) insertion, fully unrolled -> stays in VGPRs
__device__ __forceinline__ void topk_insert(float (&bd)[KNN], int (&bi)[KNN],
                                            float d, int i) {
  if (d < bd[KNN - 1]) {
    bd[KNN - 1] = d; bi[KNN - 1] = i;
#pragma unroll
    for (int j = KNN - 1; j > 0; --j) {
      if (bd[j] < bd[j - 1]) {
        float td = bd[j]; bd[j] = bd[j - 1]; bd[j - 1] = td;
        int   ti = bi[j]; bi[j] = bi[j - 1]; bi[j - 1] = ti;
      }
    }
  }
}

// ---------------------------------------------------------------------------
// Small conversion / utility kernels
// ---------------------------------------------------------------------------

__global__ void cvt_x_kernel(const float* __restrict__ x,
                             _Float16* __restrict__ xh, int total) {
  int t = blockIdx.x * 256 + threadIdx.x;
  if (t >= total) return;
#pragma unroll
  for (int c = 0; c < 4; ++c)
    xh[(size_t)t * 4 + c] = (c < 3) ? (_Float16)x[(size_t)t * 3 + c]
                                    : (_Float16)0.f;
}

__global__ void cvt_w_kernel(const float* __restrict__ W,
                             _Float16* __restrict__ Wh,
                             int O, int C, int Cpad, int total) {
  int e = blockIdx.x * 256 + threadIdx.x;
  if (e >= total) return;
  int o = e / Cpad, c = e % Cpad;
  Wh[e] = (o < O && c < C) ? (_Float16)W[(size_t)o * C + c] : (_Float16)0.f;
}

__global__ void calc_xx_kernel(const _Float16* __restrict__ Xh, int xstr,
                               int xoff, int Cin, float* __restrict__ xx,
                               int total) {
  int t = blockIdx.x * 256 + threadIdx.x;
  if (t >= total) return;
  const _Float16* p = Xh + (size_t)t * xstr + xoff;
  float s = 0.f;
  for (int c = 0; c < Cin; ++c) { float v = (float)p[c]; s += v * v; }
  xx[t] = s;
}

__global__ void zero_i32_kernel(int* __restrict__ p, int total) {
  int t = blockIdx.x * 256 + threadIdx.x;
  if (t < total) p[t] = 0;
}

// build head input (B,N,1248): [g(1024) | cat(192) | onehot(16) | pad(16)]
__global__ void fill_h_kernel(const int* __restrict__ g,
                              const _Float16* __restrict__ cat,
                              const float* __restrict__ onehot,
                              _Float16* __restrict__ HH, int N, long total) {
  long t = (long)blockIdx.x * 256 + threadIdx.x;
  if (t >= total) return;
  int c = (int)(t % 1248);
  long bn = t / 1248;
  int b = (int)(bn / N);
  _Float16 v = (_Float16)0.f;
  if (c < 1024)      v = (_Float16)__int_as_float(g[b * 1024 + c]);
  else if (c < 1216) v = cat[bn * 192 + (c - 1024)];
  else if (c < 1232) v = (_Float16)onehot[b * 16 + (c - 1216)];
  HH[t] = v;
}

// ---------------------------------------------------------------------------
// kNN on raw coords (C=3): 1 thread per query, LDS candidate tiles
// ---------------------------------------------------------------------------
__global__ void __launch_bounds__(128)
knn_small_kernel(const float* __restrict__ x, int* __restrict__ idxOut, int N) {
  __shared__ float cp[128 * 3];
  int b = blockIdx.y;
  int q = blockIdx.x * 128 + threadIdx.x;
  const float* qp = x + ((size_t)b * N + q) * 3;
  float qx = qp[0], qy = qp[1], qz = qp[2];
  float bd[KNN]; int bi[KNN];
#pragma unroll
  for (int j = 0; j < KNN; ++j) { bd[j] = 3.0e38f; bi[j] = 0; }
  for (int c0 = 0; c0 < N; c0 += 128) {
    __syncthreads();
    const float* sp = x + ((size_t)b * N + c0 + threadIdx.x) * 3;
    cp[threadIdx.x * 3 + 0] = sp[0];
    cp[threadIdx.x * 3 + 1] = sp[1];
    cp[threadIdx.x * 3 + 2] = sp[2];
    __syncthreads();
    for (int j = 0; j < 128; ++j) {
      float dx = cp[j * 3 + 0] - qx;
      float dy = cp[j * 3 + 1] - qy;
      float dz = cp[j * 3 + 2] - qz;
      topk_insert(bd, bi, dx * dx + dy * dy + dz * dz, c0 + j);
    }
  }
#pragma unroll
  for (int j = 0; j < KNN; ++j)
    idxOut[((size_t)b * N + q) * KNN + j] = bi[j];
}

// ---------------------------------------------------------------------------
// kNN on 64-ch f16 features via WMMA inner products.
// One wave handles 32 queries; score = ||c||^2 - 2 q.c (query-norm dropped:
// constant per query, never changes the ranking).
// ---------------------------------------------------------------------------
__global__ void __launch_bounds__(32)
knn_wmma_kernel(const _Float16* __restrict__ Xh, int cstride, int coff,
                const float* __restrict__ xx, int* __restrict__ idxOut,
                int N) {
  __shared__ float dtile[2][16][16];
  int b = blockIdx.y;
  int q0 = blockIdx.x * 32;
  int lane = threadIdx.x;
  int half = lane >> 4, col = lane & 15;

  v16h Aop[2][2];
#pragma unroll
  for (int ms = 0; ms < 2; ++ms)
#pragma unroll
    for (int ks = 0; ks < 2; ++ks) {
      int q = q0 + ms * 16 + col;
      Aop[ms][ks] =
          load_a16(Xh + ((size_t)b * N + q) * cstride + coff + ks * 32, half);
    }

  float bd[KNN]; int bi[KNN];
#pragma unroll
  for (int j = 0; j < KNN; ++j) { bd[j] = 3.0e38f; bi[j] = 0; }

  for (int c0 = 0; c0 < N; c0 += 16) {
    v8f acc0 = {}; v8f acc1 = {};
#pragma unroll
    for (int ks = 0; ks < 2; ++ks) {
      int cpt = c0 + col;
      v16h Bop = load_b16(Xh + ((size_t)b * N + cpt) * cstride + coff +
                          ks * 32 + (half ? 16 : 0));
      acc0 = wmma16(Aop[0][ks], Bop, acc0);
      acc1 = wmma16(Aop[1][ks], Bop, acc1);
    }
    float xc = xx[(size_t)b * N + c0 + col];
#pragma unroll
    for (int vg = 0; vg < 8; ++vg) {
      dtile[0][vg + 8 * half][col] = xc - 2.f * acc0[vg];
      dtile[1][vg + 8 * half][col] = xc - 2.f * acc1[vg];
    }
    __syncthreads();
#pragma unroll
    for (int j = 0; j < 16; ++j)
      topk_insert(bd, bi, dtile[lane >> 4][lane & 15][j], c0 + j);
    __syncthreads();
  }
  int q = q0 + lane;
#pragma unroll
  for (int j = 0; j < KNN; ++j)
    idxOut[((size_t)b * N + q) * KNN + j] = bi[j];
}

// ---------------------------------------------------------------------------
// Fused edge-conv block: gather edge features -> WMMA conv (+optional 2nd
// conv) with BN+ReLU -> masked max over k -> f16 output column slice.
// Block: 128 threads (4 waves), 16 points; k padded 20 -> 32 rows/point.
// ---------------------------------------------------------------------------
__device__ __forceinline__ void layer_mm(const _Float16* A, int Cin,
                                         const _Float16* __restrict__ W,
                                         const float* __restrict__ bb,
                                         const float* __restrict__ ss,
                                         const float* __restrict__ tt,
                                         _Float16* Ydst, int* omax, int tid) {
  int wv = tid >> 5, lane = tid & 31, half = lane >> 4, col = lane & 15;
  int ksteps = Cin >> 5;
  for (int mi = 0; mi < 8; ++mi) {
    int mt = wv * 8 + mi;
    int r = mt * 16 + col;                       // A-fragment row for this lane
#pragma unroll
    for (int nt = 0; nt < 4; ++nt) {
      v8f acc = {};
      for (int ks = 0; ks < ksteps; ++ks) {
        v16h a  = load_a16(A + (size_t)r * Cin + ks * 32, half);
        v16h bm = load_b16(W + (size_t)(nt * 16 + col) * Cin + ks * 32 +
                           (half ? 16 : 0));
        acc = wmma16(a, bm, acc);
      }
      int o = nt * 16 + col;
      float s_ = ss[o], t_ = tt[o], b_ = bb[o];
      if (Ydst) {
#pragma unroll
        for (int vg = 0; vg < 8; ++vg) {
          int rr = mt * 16 + vg + 8 * half;
          float y = fmaxf((acc[vg] + b_) * s_ + t_, 0.f);
          Ydst[(size_t)rr * 64 + o] = (_Float16)y;
        }
      } else {
        // masked max over k rows (post-ReLU >= 0 -> int-bit max valid)
        int krbase = (mt & 1) * 16 + 8 * half;
        float m = 0.f; bool any = false;
#pragma unroll
        for (int vg = 0; vg < 8; ++vg) {
          if (krbase + vg < KNN) {
            float y = fmaxf((acc[vg] + b_) * s_ + t_, 0.f);
            m = fmaxf(m, y); any = true;
          }
        }
        if (any) atomicMax(&omax[(mt >> 1) * 64 + o], __float_as_int(m));
      }
    }
  }
}

__global__ void __launch_bounds__(128)
edgeconv_kernel(const _Float16* __restrict__ src, int sstr, int soff, int Csrc,
                int CpadE, const int* __restrict__ idx,
                const _Float16* __restrict__ Wa, const float* __restrict__ ba,
                const float* __restrict__ sa, const float* __restrict__ ta,
                const _Float16* __restrict__ Wb, const float* __restrict__ bbp,
                const float* __restrict__ sb, const float* __restrict__ tb,
                int hasB, _Float16* __restrict__ dst, int dstr, int doff,
                int N) {
  extern __shared__ char smem[];
  _Float16* E  = (_Float16*)smem;                    // 16*32*CpadE
  _Float16* Y1 = E + (size_t)16 * 32 * CpadE;        // 16*32*64 if hasB
  int* omax = (int*)(Y1 + (hasB ? (size_t)16 * 32 * 64 : 0));  // 16*64

  int b = blockIdx.y, n0 = blockIdx.x * 16, tid = threadIdx.x;
  for (int e = tid; e < 16 * 64; e += 128) omax[e] = 0;

  // gather edge features: rows r = p*32 + kr, cols [xj-xi | xi | 0-pad]
  for (int e = tid; e < 16 * 32 * CpadE; e += 128) {
    int c = e % CpadE, r = e / CpadE, kr = r & 31, p = r >> 5;
    _Float16 v = (_Float16)0.f;
    if (kr < KNN) {
      int nq = n0 + p;
      int j = idx[((size_t)b * N + nq) * KNN + kr];
      const _Float16* cs = src + ((size_t)b * N + nq) * sstr + soff;
      const _Float16* js = src + ((size_t)b * N + j) * sstr + soff;
      if (c < Csrc)            v = (_Float16)((float)js[c] - (float)cs[c]);
      else if (c < 2 * Csrc)   v = cs[c - Csrc];
    }
    E[e] = v;
  }
  __syncthreads();

  layer_mm(E, CpadE, Wa, ba, sa, ta, hasB ? Y1 : nullptr, omax, tid);
  __syncthreads();
  if (hasB) {
    layer_mm(Y1, 64, Wb, bbp, sb, tb, nullptr, omax, tid);
    __syncthreads();
  }
  for (int e = tid; e < 16 * 64; e += 128) {
    int p = e >> 6, o = e & 63;
    dst[((size_t)b * N + n0 + p) * dstr + doff + o] =
        (_Float16)__int_as_float(omax[e]);
  }
}

// ---------------------------------------------------------------------------
// Pointwise GEMM: Y = act(W*X + b). 128 threads = 4 waves, 16 points x 64
// outputs per block. The 16 x Cpad activation tile is staged to LDS once by
// the Tensor Data Mover (2D D#: tile 16 rows x Cpad elems, stride xstr) and
// shared by all 4 waves. Output modes:
//   gmax != 0 : global max-over-points reduction (lin1)
//   Yf   != 0 : f32 (B,N,Oreal) store (final layer -> d_out)
//   else      : f16 point-major column slice
// ---------------------------------------------------------------------------
__global__ void __launch_bounds__(128)
gemm_pw_kernel(const _Float16* __restrict__ Xh, int xstr, int xoff,
               const _Float16* __restrict__ Wh, int Cpad,
               const float* __restrict__ bb, const float* __restrict__ ss,
               const float* __restrict__ tt, int bnrelu,
               _Float16* __restrict__ Yh, int ystr, int yoff,
               float* __restrict__ Yf, int Oreal, int* __restrict__ gmax,
               int N) {
  extern __shared__ _Float16 xs[];                 // 16 x Cpad
  int b = blockIdx.z;
  int tid = threadIdx.x, wave = tid >> 5, lane = tid & 31;
  int half = lane >> 4, col = lane & 15;
  int pt0 = blockIdx.x * 16;
  int ot = blockIdx.y * 4 + wave;
  const _Float16* gx = Xh + ((size_t)b * N + pt0) * xstr + xoff;

#if USE_TDM
  if (wave == 0) {
    unsigned long long ga = (unsigned long long)(uintptr_t)gx;
    unsigned int lds = (unsigned int)(uintptr_t)xs;  // LDS byte address
    u32x4 g0;
    g0[0] = 1u;                                      // count=1 valid D#
    g0[1] = lds;                                     // lds_addr
    g0[2] = (unsigned int)(ga & 0xFFFFFFFFu);        // global_addr[31:0]
    g0[3] = (unsigned int)((ga >> 32) & 0x1FFFFFFu)  // global_addr[56:32]
            | (2u << 30);                            // type=2 ("image")
    i32x8 g1;
    g1[0] = 1 << 16;                                 // data_size=1 (2 bytes)
    g1[1] = (int)(((unsigned)Cpad & 0xFFFFu) << 16); // tensor_dim0[15:0]
    g1[2] = (int)(((unsigned)Cpad >> 16) | (16u << 16)); // dim0 hi | dim1=16
    g1[3] = (int)((unsigned)Cpad << 16);             // dim1 hi=0 | tile_dim0
    g1[4] = 16;                                      // tile_dim1=16, dim2=0
    g1[5] = xstr;                                    // dim0_stride[31:0]
    g1[6] = 0;                                       // stride hi / dim1_stride
    g1[7] = 0;
    i32x4 z4 = {0, 0, 0, 0};
    i32x8 z8 = {0, 0, 0, 0, 0, 0, 0, 0};
    __builtin_amdgcn_tensor_load_to_lds(g0, g1, z4, z4, z8, 0);
    __builtin_amdgcn_s_wait_tensorcnt(0);
  }
  __syncthreads();
#else
  for (int e = tid; e < 16 * Cpad; e += 128)
    xs[e] = gx[(size_t)(e / Cpad) * xstr + (e % Cpad)];
  __syncthreads();
#endif

  const _Float16* arow = Wh + (size_t)(ot * 16 + col) * Cpad;
  const _Float16* xrow = xs + (size_t)col * Cpad + (half ? 16 : 0);
  int ksteps = Cpad >> 5;
  v8f acc = {};
  for (int ks = 0; ks < ksteps; ++ks) {
    v16h a  = load_a16(arow + ks * 32, half);
    v16h bm = load_b16(xrow + ks * 32);
    __builtin_prefetch(arow + (ks + 1) * 32, 0, 1);
    acc = wmma16(a, bm, acc);
  }
#pragma unroll
  for (int vg = 0; vg < 8; ++vg) {
    int o = ot * 16 + vg + 8 * half;
    int pt = pt0 + col;
    float y = acc[vg] + bb[o];
    if (bnrelu) y = fmaxf(y * ss[o] + tt[o], 0.f);
    if (gmax) {
      float m = y;
      m = fmaxf(m, __shfl_xor(m, 1, 32));
      m = fmaxf(m, __shfl_xor(m, 2, 32));
      m = fmaxf(m, __shfl_xor(m, 4, 32));
      m = fmaxf(m, __shfl_xor(m, 8, 32));
      if (col == 0) atomicMax(&gmax[b * 1024 + o], __float_as_int(m));
    } else if (Yf) {
      if (o < Oreal) Yf[((size_t)b * N + pt) * Oreal + o] = y;
    } else {
      Yh[((size_t)b * N + pt) * ystr + yoff + o] = (_Float16)y;
    }
  }
}

// ---------------------------------------------------------------------------
// Host orchestration
// ---------------------------------------------------------------------------
extern "C" void kernel_launch(void* const* d_in, const int* in_sizes, int n_in,
                              void* d_out, int out_size, void* d_ws,
                              size_t ws_size, hipStream_t stream) {
  const float* x      = (const float*)d_in[0];
  const float* onehot = (const float*)d_in[1];
  const float *c1aw = (const float*)d_in[2],  *c1ab = (const float*)d_in[3],
              *c1as = (const float*)d_in[4],  *c1at = (const float*)d_in[5];
  const float *c1bw = (const float*)d_in[6],  *c1bb = (const float*)d_in[7],
              *c1bs = (const float*)d_in[8],  *c1bt = (const float*)d_in[9];
  const float *c2aw = (const float*)d_in[10], *c2ab = (const float*)d_in[11],
              *c2as = (const float*)d_in[12], *c2at = (const float*)d_in[13];
  const float *c2bw = (const float*)d_in[14], *c2bb = (const float*)d_in[15],
              *c2bs = (const float*)d_in[16], *c2bt = (const float*)d_in[17];
  const float *c3aw = (const float*)d_in[18], *c3ab = (const float*)d_in[19],
              *c3as = (const float*)d_in[20], *c3at = (const float*)d_in[21];
  const float *l1w  = (const float*)d_in[22], *l1b  = (const float*)d_in[23],
              *l1s  = (const float*)d_in[24], *l1t  = (const float*)d_in[25];
  const float *k1w  = (const float*)d_in[26], *k1b  = (const float*)d_in[27],
              *k1s  = (const float*)d_in[28], *k1t  = (const float*)d_in[29];
  const float *k2w  = (const float*)d_in[30], *k2b  = (const float*)d_in[31],
              *k2s  = (const float*)d_in[32], *k2t  = (const float*)d_in[33];
  const float *k3w  = (const float*)d_in[34], *k3b  = (const float*)d_in[35],
              *k3s  = (const float*)d_in[36], *k3t  = (const float*)d_in[37];
  const float *k4w  = (const float*)d_in[38], *k4b  = (const float*)d_in[39];

  const int B = in_sizes[1] / 16;
  const int N = in_sizes[0] / (3 * B);

  char* wp = (char*)d_ws;
  auto alloc = [&](size_t bytes) -> void* {
    void* p = (void*)wp;
    wp += (bytes + 255) & ~(size_t)255;
    return p;
  };

  _Float16* X3h  = (_Float16*)alloc((size_t)B * N * 4 * 2);
  int*      idx  = (int*)alloc((size_t)B * N * KNN * 4);
  float*    xx   = (float*)alloc((size_t)B * N * 4);
  _Float16* CATH = (_Float16*)alloc((size_t)B * N * 192 * 2);
  int*      g    = (int*)alloc((size_t)B * 1024 * 4);
  _Float16* HH   = (_Float16*)alloc((size_t)B * N * 1248 * 2);
  _Float16* H1   = (_Float16*)alloc((size_t)B * N * 256 * 2);
  _Float16* H2   = (_Float16*)alloc((size_t)B * N * 256 * 2);
  _Float16* H3   = (_Float16*)alloc((size_t)B * N * 128 * 2);
  _Float16* Wc1a = (_Float16*)alloc((size_t)64 * 32 * 2);
  _Float16* Wc1b = (_Float16*)alloc((size_t)64 * 64 * 2);
  _Float16* Wc2a = (_Float16*)alloc((size_t)64 * 128 * 2);
  _Float16* Wc2b = (_Float16*)alloc((size_t)64 * 64 * 2);
  _Float16* Wc3a = (_Float16*)alloc((size_t)64 * 128 * 2);
  _Float16* Wl1  = (_Float16*)alloc((size_t)1024 * 192 * 2);
  _Float16* Wk1  = (_Float16*)alloc((size_t)256 * 1248 * 2);
  _Float16* Wk2  = (_Float16*)alloc((size_t)256 * 256 * 2);
  _Float16* Wk3  = (_Float16*)alloc((size_t)128 * 256 * 2);
  _Float16* Wk4  = (_Float16*)alloc((size_t)64 * 128 * 2);

  auto cvtw = [&](const float* W, _Float16* Wh, int O, int C, int Opad,
                  int Cpad) {
    int total = Opad * Cpad;
    cvt_w_kernel<<<(total + 255) / 256, 256, 0, stream>>>(W, Wh, O, C, Cpad,
                                                          total);
  };
  cvtw(c1aw, Wc1a, 64, 6, 64, 32);
  cvtw(c1bw, Wc1b, 64, 64, 64, 64);
  cvtw(c2aw, Wc2a, 64, 128, 64, 128);
  cvtw(c2bw, Wc2b, 64, 64, 64, 64);
  cvtw(c3aw, Wc3a, 64, 128, 64, 128);
  cvtw(l1w, Wl1, 1024, 192, 1024, 192);
  cvtw(k1w, Wk1, 256, 1232, 256, 1248);
  cvtw(k2w, Wk2, 256, 256, 256, 256);
  cvtw(k3w, Wk3, 128, 256, 128, 256);
  cvtw(k4w, Wk4, 50, 128, 64, 128);

  cvt_x_kernel<<<(B * N + 255) / 256, 256, 0, stream>>>(x, X3h, B * N);

  // ---- stage 1: knn(C=3) + edgeconv(6->64->64) -> CATH[:,0:64]
  knn_small_kernel<<<dim3(N / 128, B), 128, 0, stream>>>(x, idx, N);
  {
    size_t sh = (size_t)16 * 32 * 32 * 2 + (size_t)16 * 32 * 64 * 2 +
                (size_t)16 * 64 * 4;
    edgeconv_kernel<<<dim3(N / 16, B), 128, sh, stream>>>(
        X3h, 4, 0, 3, 32, idx, Wc1a, c1ab, c1as, c1at, Wc1b, c1bb, c1bs, c1bt,
        1, CATH, 192, 0, N);
  }
  // ---- stage 2: knn(C=64, WMMA) + edgeconv(128->64->64) -> CATH[:,64:128]
  calc_xx_kernel<<<(B * N + 255) / 256, 256, 0, stream>>>(CATH, 192, 0, 64, xx,
                                                          B * N);
  knn_wmma_kernel<<<dim3(N / 32, B), 32, 0, stream>>>(CATH, 192, 0, xx, idx, N);
  {
    size_t sh = (size_t)16 * 32 * 128 * 2 + (size_t)16 * 32 * 64 * 2 +
                (size_t)16 * 64 * 4;
    edgeconv_kernel<<<dim3(N / 16, B), 128, sh, stream>>>(
        CATH, 192, 0, 64, 128, idx, Wc2a, c2ab, c2as, c2at, Wc2b, c2bb, c2bs,
        c2bt, 1, CATH, 192, 64, N);
  }
  // ---- stage 3: knn(C=64) + edgeconv(128->64) -> CATH[:,128:192]
  calc_xx_kernel<<<(B * N + 255) / 256, 256, 0, stream>>>(CATH, 192, 64, 64,
                                                          xx, B * N);
  knn_wmma_kernel<<<dim3(N / 32, B), 32, 0, stream>>>(CATH, 192, 64, xx, idx,
                                                      N);
  {
    size_t sh = (size_t)16 * 32 * 128 * 2 + (size_t)16 * 64 * 4;
    edgeconv_kernel<<<dim3(N / 16, B), 128, sh, stream>>>(
        CATH, 192, 0, 64, 128, idx, Wc3a, c3ab, c3as, c3at, nullptr, nullptr,
        nullptr, nullptr, 0, CATH, 192, 128, N);
  }

  // ---- lin1 (192->1024) with fused global max over points -> g
  zero_i32_kernel<<<(B * 1024 + 255) / 256, 256, 0, stream>>>(g, B * 1024);
  gemm_pw_kernel<<<dim3(N / 16, 1024 / 64, B), 128, (size_t)16 * 192 * 2,
                   stream>>>(CATH, 192, 0, Wl1, 192, l1b, l1s, l1t, 1, nullptr,
                             0, 0, nullptr, 0, g, N);

  // ---- head input (B,N,1248) = [g | x1x2x3 | onehot | pad]
  {
    long total = (long)B * N * 1248;
    fill_h_kernel<<<(int)((total + 255) / 256), 256, 0, stream>>>(
        g, CATH, onehot, HH, N, total);
  }

  // ---- classifier chain
  gemm_pw_kernel<<<dim3(N / 16, 256 / 64, B), 128, (size_t)16 * 1248 * 2,
                   stream>>>(HH, 1248, 0, Wk1, 1248, k1b, k1s, k1t, 1, H1, 256,
                             0, nullptr, 0, nullptr, N);
  gemm_pw_kernel<<<dim3(N / 16, 256 / 64, B), 128, (size_t)16 * 256 * 2,
                   stream>>>(H1, 256, 0, Wk2, 256, k2b, k2s, k2t, 1, H2, 256,
                             0, nullptr, 0, nullptr, N);
  gemm_pw_kernel<<<dim3(N / 16, 128 / 64, B), 128, (size_t)16 * 256 * 2,
                   stream>>>(H2, 256, 0, Wk3, 256, k3b, k3s, k3t, 1, H3, 128,
                             0, nullptr, 0, nullptr, N);
  gemm_pw_kernel<<<dim3(N / 16, 64 / 64, B), 128, (size_t)16 * 128 * 2,
                   stream>>>(H3, 128, 0, Wk4, 128, k4b, nullptr, nullptr, 0,
                             nullptr, 0, 0, (float*)d_out, 50, nullptr, N);
}
